// multi_attention_33492154974632
// MI455X (gfx1250) — matrix operand order repaired and checked by reading, not compiled, not run on hardware
//
#include <hip/hip_runtime.h>
#include <hip/hip_bf16.h>

// ---------------------------------------------------------------------------
// Types
// ---------------------------------------------------------------------------
typedef __bf16 bf16_t;
typedef __attribute__((ext_vector_type(16))) __bf16 v16bf;
typedef __attribute__((ext_vector_type(8)))  __bf16 v8bf;
typedef __attribute__((ext_vector_type(8)))  float  v8f;

static __device__ inline v16bf cat8(v8bf lo, v8bf hi) {
  return __builtin_shufflevector(lo, hi, 0,1,2,3,4,5,6,7,8,9,10,11,12,13,14,15);
}
static __device__ inline v8f zero_v8f() {
  v8f z = {0.f,0.f,0.f,0.f,0.f,0.f,0.f,0.f};
  return z;
}
// float -> bf16, round-to-nearest-even (bit math; no reliance on __bf16 arith)
static __device__ inline bf16_t f2b(float f) {
  unsigned u = __float_as_uint(f);
  u = (u + 0x7FFFu + ((u >> 16) & 1u)) >> 16;
  unsigned short us = (unsigned short)u;
  bf16_t b;
  __builtin_memcpy(&b, &us, 2);
  return b;
}
static __device__ inline float b2f(bf16_t b) {
  unsigned short us;
  __builtin_memcpy(&us, &b, 2);
  return __uint_as_float(((unsigned)us) << 16);
}
static __device__ inline unsigned pack2(float a, float b) {
  bf16_t ba = f2b(a), bb = f2b(b);
  unsigned short ua, ub;
  __builtin_memcpy(&ua, &ba, 2);
  __builtin_memcpy(&ub, &bb, 2);
  return (unsigned)ua | ((unsigned)ub << 16);
}

// CDNA5 async memory->LDS (ASYNCcnt-tracked, no VGPR round trip).
// Generic pointer to LDS truncated to 32 bits == wave-relative LDS offset.
static __device__ inline void async_ld_b128(void* lds, const void* gaddr) {
  asm volatile("global_load_async_to_lds_b128 %0, %1, off"
               :: "v"((unsigned)(unsigned long long)lds), "v"(gaddr)
               : "memory");
}
// Wait until at most N async ops outstanding (in-order retirement => the
// oldest issued batch has landed in LDS).
template<int N>
static __device__ inline void wait_async() {
  asm volatile("s_wait_asynccnt %0" :: "i"(N) : "memory");
}

// Problem constants
#define BQ   4
#define TT   2048
#define CC   1024
#define HH   16
#define DH   64

// ---------------------------------------------------------------------------
// K0: fp32 -> bf16 conversion
// ---------------------------------------------------------------------------
__global__ __launch_bounds__(256) void f2bf_kernel(const float* __restrict__ in,
                                                   bf16_t* __restrict__ out, int n) {
  int i = blockIdx.x * 256 + threadIdx.x;
  if (i < n) out[i] = f2b(in[i]);
}

// ---------------------------------------------------------------------------
// K1/K4: GEMM  C[M,N] = A[M,K] * B[N,K]^T   (bf16 in, f32 accum)
//   block tile 128x128, k-step 32, 8 wave32 waves (2x4 wave grid),
//   each wave 64x32 via 4x2 v_wmma_f32_16x16x32_bf16 tiles.
//   Double-buffered LDS tiles streamed with global_load_async_to_lds_b128;
//   s_wait_asynccnt 4 keeps the next tile in flight behind the WMMAs.
// ---------------------------------------------------------------------------
template<bool OUT_BF16>
__global__ __launch_bounds__(256) void gemm_xwt_bf16(
    const bf16_t* __restrict__ A, const bf16_t* __restrict__ B,
    void* __restrict__ Cout, int M, int N, int K)
{
  __shared__ bf16_t sA[2][128 * 40];   // 128 rows x 32 K, padded stride 40
  __shared__ bf16_t sB[2][128 * 40];

  const int tid   = threadIdx.x;
  const int lane  = tid & 31;
  const int wave  = tid >> 5;
  const int wm    = wave & 1;       // 2 waves along M
  const int wn    = wave >> 1;      // 4 waves along N
  const int khalf = lane >> 4;
  const int lr    = lane & 15;
  const int blockM = blockIdx.y * 128;
  const int blockN = blockIdx.x * 128;
  const int nk    = K >> 5;

  auto issue_tile = [&](int kt, int buf) {
    #pragma unroll
    for (int c = 0; c < 2; ++c) {
      int cid = tid + c * 256;
      int row = cid >> 2, ch = cid & 3;
      async_ld_b128(&sA[buf][row * 40 + ch * 8],
                    &A[(size_t)(blockM + row) * K + kt * 32 + ch * 8]);
      async_ld_b128(&sB[buf][row * 40 + ch * 8],
                    &B[(size_t)(blockN + row) * K + kt * 32 + ch * 8]);
    }
  };

  v8f acc[4][2];
  #pragma unroll
  for (int i = 0; i < 4; ++i) { acc[i][0] = zero_v8f(); acc[i][1] = zero_v8f(); }

  issue_tile(0, 0);
  for (int kt = 0; kt < nk; ++kt) {
    const int buf = kt & 1;
    __syncthreads();                 // readers of buf^1 (iter kt-1) are done
    if (kt + 1 < nk) {
      issue_tile(kt + 1, buf ^ 1);   // stream next tile behind compute
      wait_async<4>();               // tile kt resident; kt+1 in flight
      // L2 warm-up two tiles ahead
      int row = tid >> 2, ch = tid & 3;
      if (kt + 2 < nk) {
        __builtin_prefetch(&A[(size_t)(blockM + row) * K + (kt + 2) * 32 + ch * 8], 0, 0);
        __builtin_prefetch(&B[(size_t)(blockN + row) * K + (kt + 2) * 32 + ch * 8], 0, 0);
      }
    } else {
      wait_async<0>();
    }
    __syncthreads();                 // tile kt visible to all waves

    // A fragments: 16x32 bf16; lane m = lr, lo8 @ khalf*8, hi8 @ 16+khalf*8
    v16bf af[4];
    #pragma unroll
    for (int mt = 0; mt < 4; ++mt) {
      const bf16_t* pa = &sA[buf][(wm * 64 + mt * 16 + lr) * 40];
      af[mt] = cat8(*(const v8bf*)(pa + khalf * 8),
                    *(const v8bf*)(pa + 16 + khalf * 8));
    }
    // B fragments: 32x16; lane n = lr, lo8 @ khalf*8, hi8 @ 16+khalf*8
    #pragma unroll
    for (int nt = 0; nt < 2; ++nt) {
      const bf16_t* pb = &sB[buf][(wn * 32 + nt * 16 + lr) * 40 + khalf * 8];
      v16bf bfg = cat8(*(const v8bf*)pb, *(const v8bf*)(pb + 16));
      #pragma unroll
      for (int mt = 0; mt < 4; ++mt) {
        acc[mt][nt] = __builtin_amdgcn_wmma_f32_16x16x32_bf16(
            false, af[mt], false, bfg, (short)0, acc[mt][nt], false, false);
      }
    }
  }

  // Epilogue: C layout -> row = r + 8*khalf, col = lr
  #pragma unroll
  for (int mt = 0; mt < 4; ++mt) {
    #pragma unroll
    for (int nt = 0; nt < 2; ++nt) {
      #pragma unroll
      for (int r = 0; r < 8; ++r) {
        int row = blockM + wm * 64 + mt * 16 + r + 8 * khalf;
        int col = blockN + wn * 32 + nt * 16 + lr;
        float v = acc[mt][nt][r];
        if (OUT_BF16) ((bf16_t*)Cout)[(size_t)row * N + col] = f2b(v);
        else          ((float*) Cout)[(size_t)row * N + col] = v;
      }
    }
  }
}

// ---------------------------------------------------------------------------
// K2: RoPE + head split with LDS-tiled V transpose.
//   One block per (b, h, 64-row t-tile).  qkv[B*T, 3C] (bf16) ->
//     q_r[b,h,t,d], k_r[b,h,t,d]  (roped, lane-contiguous packed stores)
//     v_t[b,h,d,t]                (transposed via LDS, 16B-coalesced both ways)
// ---------------------------------------------------------------------------
__global__ __launch_bounds__(256) void rope_split_kernel(
    const bf16_t* __restrict__ qkv, bf16_t* __restrict__ q_r,
    bf16_t* __restrict__ k_r, bf16_t* __restrict__ v_t)
{
  __shared__ bf16_t sT[64 * 72];     // 64 t-rows x 64 d, padded stride 72

  const int tid   = threadIdx.x;
  const int ttile = blockIdx.x & 31;
  const int h     = (blockIdx.x >> 5) & 15;
  const int b     = blockIdx.x >> 9;
  const int t0    = ttile * 64;

  // ---- Phase 1: RoPE on Q and K. 64 t x 32 pairs = 2048 tasks, 8/thread.
  #pragma unroll
  for (int c = 0; c < 8; ++c) {
    int id = c * 256 + tid;
    int i  = id & 31;          // pair index (varies fastest -> coalesced)
    int tl = id >> 5;          // 0..63
    int t  = t0 + tl;
    const bf16_t* row = qkv + (size_t)(b * TT + t) * (3 * CC);

    float inv = __expf(-((float)(2 * i) / 64.f) * 9.210340371976184f);
    float ang = (float)t * inv;
    float s, co;
    __sincosf(ang, &s, &co);

    size_t qo = ((size_t)(b * HH + h) * TT + t) * DH + 2 * i;
    float x1 = b2f(row[h * DH + 2 * i]);
    float x2 = b2f(row[h * DH + 2 * i + 1]);
    *(unsigned*)&q_r[qo] = pack2(x1 * co - x2 * s, x1 * s + x2 * co);
    x1 = b2f(row[CC + h * DH + 2 * i]);
    x2 = b2f(row[CC + h * DH + 2 * i + 1]);
    *(unsigned*)&k_r[qo] = pack2(x1 * co - x2 * s, x1 * s + x2 * co);
  }

  // ---- Phase 2: V transpose through LDS.
  // Load 64 t-rows x 64 d (16B chunks, coalesced): 512 chunks, 2/thread.
  #pragma unroll
  for (int c = 0; c < 2; ++c) {
    int id = c * 256 + tid;
    int tl = id >> 3, ch = id & 7;
    *(v8bf*)&sT[tl * 72 + ch * 8] =
        *(const v8bf*)&qkv[(size_t)(b * TT + t0 + tl) * (3 * CC)
                           + 2 * CC + h * DH + ch * 8];
  }
  __syncthreads();
  // Store 64 d-rows x 64 t (16B chunks, coalesced): 512 chunks, 2/thread.
  #pragma unroll
  for (int c = 0; c < 2; ++c) {
    int id = c * 256 + tid;
    int d = id >> 3, ch = id & 7;
    v8bf tmp;
    #pragma unroll
    for (int e = 0; e < 8; ++e) tmp[e] = sT[(ch * 8 + e) * 72 + d];
    *(v8bf*)&v_t[((size_t)(b * HH + h) * DH + d) * TT + t0 + ch * 8] = tmp;
  }
}

// ---------------------------------------------------------------------------
// K3: causal flash attention, bf16 WMMA, f32 online softmax.
//   grid = (T/128, B*H); 256 threads = 8 waves; wave w owns q rows
//   [qc*128 + w*16, +16). Double-buffered K/V tiles (32 keys) async-staged
//   in LDS and shared by all waves. att out: [b,t,h*64+d] bf16.
// ---------------------------------------------------------------------------
__global__ __launch_bounds__(256) void flash_attn_kernel(
    const bf16_t* __restrict__ q_r, const bf16_t* __restrict__ k_r,
    const bf16_t* __restrict__ v_t, bf16_t* __restrict__ att)
{
  __shared__ bf16_t sK[2][32 * 72];    // 32 keys x 64 d, stride 72
  __shared__ bf16_t sV[2][64 * 40];    // 64 d x 32 keys, stride 40
  __shared__ bf16_t sP[8][16 * 40];    // per-wave P relayout scratch

  const int tid   = threadIdx.x;
  const int lane  = tid & 31;
  const int wave  = tid >> 5;
  const int khalf = lane >> 4;
  const int lr    = lane & 15;
  const int bh    = blockIdx.y;
  const int qc    = blockIdx.x;
  const int q0    = qc * 128 + wave * 16;

  const bf16_t* qBase = q_r + (size_t)bh * TT * DH;
  const bf16_t* kBase = k_r + (size_t)bh * TT * DH;
  const bf16_t* vBase = v_t + (size_t)bh * DH * TT;

  auto issue_kv = [&](int kt, int buf) {
    int key = tid >> 3, ch = tid & 7;
    async_ld_b128(&sK[buf][key * 72 + ch * 8],
                  &kBase[(size_t)(kt * 32 + key) * DH + ch * 8]);
    int d = tid >> 2, ch2 = tid & 3;
    async_ld_b128(&sV[buf][d * 40 + ch2 * 8],
                  &vBase[(size_t)d * TT + kt * 32 + ch2 * 8]);
  };

  // Q fragments (A layout), kept in registers for whole key loop
  v16bf qf[2];
  #pragma unroll
  for (int ks = 0; ks < 2; ++ks) {
    const bf16_t* p = qBase + (size_t)(q0 + lr) * DH + ks * 32;
    qf[ks] = cat8(*(const v8bf*)(p + khalf * 8),
                  *(const v8bf*)(p + 16 + khalf * 8));
  }

  v8f o[4];
  #pragma unroll
  for (int j = 0; j < 4; ++j) o[j] = zero_v8f();
  float m_i[8], l_i[8];
  #pragma unroll
  for (int r = 0; r < 8; ++r) { m_i[r] = -__builtin_inff(); l_i[r] = 0.f; }

  const float scale = 0.125f;               // 1/sqrt(64)
  const int nkt = qc * 4 + 4;               // causal bound for this block

  issue_kv(0, 0);
  for (int kt = 0; kt < nkt; ++kt) {
    const int buf = kt & 1;
    __syncthreads();                 // readers of buf^1 are done
    if (kt + 1 < nkt) {
      issue_kv(kt + 1, buf ^ 1);     // stream next K/V tile behind compute
      wait_async<2>();               // tile kt resident; kt+1 in flight
    } else {
      wait_async<0>();
    }
    __syncthreads();

    // S = Q K^T : two 16x16 tiles (j = key half)
    v8f sfr[2];
    #pragma unroll
    for (int j = 0; j < 2; ++j) {
      sfr[j] = zero_v8f();
      #pragma unroll
      for (int ks = 0; ks < 2; ++ks) {
        const bf16_t* pb = &sK[buf][(j * 16 + lr) * 72 + ks * 32 + khalf * 8];
        v16bf bk = cat8(*(const v8bf*)pb, *(const v8bf*)(pb + 16));
        sfr[j] = __builtin_amdgcn_wmma_f32_16x16x32_bf16(
            false, qf[ks], false, bk, (short)0, sfr[j], false, false);
      }
    }

    // Online softmax per row (row = r + 8*khalf; cols across 16-lane group)
    int key0 = kt * 32 + lr;
    float fac[8];
    #pragma unroll
    for (int r = 0; r < 8; ++r) {
      int qrow = q0 + r + 8 * khalf;
      float v0 = (key0      <= qrow) ? sfr[0][r] * scale : -__builtin_inff();
      float v1 = (key0 + 16 <= qrow) ? sfr[1][r] * scale : -__builtin_inff();
      float mt = fmaxf(v0, v1);
      #pragma unroll
      for (int off = 1; off < 16; off <<= 1) mt = fmaxf(mt, __shfl_xor(mt, off));
      float mn = fmaxf(m_i[r], mt);
      float p0 = __expf(v0 - mn);
      float p1 = __expf(v1 - mn);
      float rs = p0 + p1;
      #pragma unroll
      for (int off = 1; off < 16; off <<= 1) rs += __shfl_xor(rs, off);
      float f = __expf(m_i[r] - mn);
      l_i[r] = l_i[r] * f + rs;
      m_i[r] = mn;
      fac[r] = f;
      sP[wave][(r + 8 * khalf) * 40 + lr]      = f2b(p0);
      sP[wave][(r + 8 * khalf) * 40 + 16 + lr] = f2b(p1);
    }
    // Rescale O
    #pragma unroll
    for (int j = 0; j < 4; ++j)
      #pragma unroll
      for (int r = 0; r < 8; ++r) o[j][r] *= fac[r];

    // P back as A fragment (16x32), wave-local LDS round-trip
    const bf16_t* pp = &sP[wave][0];
    v16bf pf = cat8(*(const v8bf*)(pp + lr * 40 + khalf * 8),
                    *(const v8bf*)(pp + lr * 40 + 16 + khalf * 8));

    // O += P * V over 4 d-tiles
    #pragma unroll
    for (int j = 0; j < 4; ++j) {
      const bf16_t* pv = &sV[buf][(j * 16 + lr) * 40 + khalf * 8];
      v16bf bv = cat8(*(const v8bf*)pv, *(const v8bf*)(pv + 16));
      o[j] = __builtin_amdgcn_wmma_f32_16x16x32_bf16(
          false, pf, false, bv, (short)0, o[j], false, false);
    }
  }

  // Normalize and store att[b, t, h*64 + d] (bf16)
  const int b = bh >> 4, h = bh & 15;
  float rl[8];
  #pragma unroll
  for (int r = 0; r < 8; ++r) rl[r] = 1.f / l_i[r];
  #pragma unroll
  for (int j = 0; j < 4; ++j) {
    #pragma unroll
    for (int r = 0; r < 8; ++r) {
      int t = q0 + r + 8 * khalf;
      int d = j * 16 + lr;
      att[((size_t)(b * TT + t)) * CC + h * DH + d] = f2b(o[j][r] * rl[r]);
    }
  }
}

// ---------------------------------------------------------------------------
// Host launcher
// ---------------------------------------------------------------------------
extern "C" void kernel_launch(void* const* d_in, const int* in_sizes, int n_in,
                              void* d_out, int out_size, void* d_ws, size_t ws_size,
                              hipStream_t stream) {
  (void)in_sizes; (void)n_in; (void)out_size; (void)ws_size;
  const float* x      = (const float*)d_in[0];   // [4,2048,1024]
  const float* W_qkv  = (const float*)d_in[1];   // [3072,1024]
  const float* W_proj = (const float*)d_in[2];   // [1024,1024]
  float* out = (float*)d_out;                    // [4,2048,1024]

  // Workspace layout (bf16 elements)
  bf16_t* xb     = (bf16_t*)d_ws;
  bf16_t* wqkvb  = xb     + (size_t)8388608;    // x:      8192*1024
  bf16_t* wprojb = wqkvb  + (size_t)3145728;    // W_qkv:  3072*1024
  bf16_t* qkvb   = wprojb + (size_t)1048576;    // W_proj: 1024*1024
  bf16_t* q_r    = qkvb   + (size_t)25165824;   // qkv:    8192*3072
  bf16_t* k_r    = q_r    + (size_t)8388608;
  bf16_t* v_t    = k_r    + (size_t)8388608;
  bf16_t* attb   = v_t    + (size_t)8388608;

  // K0: fp32 -> bf16
  f2bf_kernel<<<32768, 256, 0, stream>>>(x,      xb,     8388608);
  f2bf_kernel<<<12288, 256, 0, stream>>>(W_qkv,  wqkvb,  3145728);
  f2bf_kernel<<< 4096, 256, 0, stream>>>(W_proj, wprojb, 1048576);

  // K1: qkv = x @ W_qkv^T   (M=8192, N=3072, K=1024), bf16 out
  gemm_xwt_bf16<true><<<dim3(24, 64), 256, 0, stream>>>(
      xb, wqkvb, (void*)qkvb, 8192, 3072, 1024);

  // K2: RoPE + head split / LDS-tiled V transpose
  rope_split_kernel<<<2048, 256, 0, stream>>>(qkvb, q_r, k_r, v_t);

  // K3: causal flash attention -> att bf16 [B,T,C]
  flash_attn_kernel<<<dim3(16, 64), 256, 0, stream>>>(q_r, k_r, v_t, attb);

  // K4: out = att @ W_proj^T  (M=8192, N=1024, K=1024), fp32 out
  gemm_xwt_bf16<false><<<dim3(8, 64), 256, 0, stream>>>(
      attb, wprojb, (void*)out, 8192, 1024, 1024);
}
